// MolecularGIN_71373766525470
// MI455X (gfx1250) — matrix operand order repaired and checked by reading, not compiled, not run on hardware
//
#include <hip/hip_runtime.h>
#include <hip/hip_bf16.h>

// ---------------------------------------------------------------------------
// MolecularGIN attention-pool forward for gfx1250 (MI455X), bf16 WMMA path.
// ---------------------------------------------------------------------------

typedef __attribute__((ext_vector_type(16))) __bf16 bf16x16;
typedef __attribute__((ext_vector_type(8)))  __bf16 bf16x8;
typedef __attribute__((ext_vector_type(8)))  float  f32x8;

#define BB   256   // batch
#define NN   128   // nodes
#define DIN  300
#define DD   768
#define KQ   256   // query tokens
#define HH   8
#define DH   96
#define NEGV (-1e9f)

// ---- WMMA fragment loaders (CDNA5 16x16x32 bf16 layouts) -------------------
// A (16x32): lane<16 -> row=lane, K = {kb..kb+7, 16+kb..16+kb+7}, kb=(lane>>4)*8
__device__ __forceinline__ bf16x16 frag_a(const __bf16* base, int ld) {
  const int lane = threadIdx.x & 31;
  const int r    = lane & 15;
  const int kb   = (lane >> 4) << 3;
  const __bf16* p = base + r * ld + kb;
  bf16x8 lo = *(const bf16x8*)(p);
  bf16x8 hi = *(const bf16x8*)(p + 16);
  bf16x16 f;
#pragma unroll
  for (int i = 0; i < 8; ++i) { f[i] = lo[i]; f[i + 8] = hi[i]; }
  return f;
}
// B (32x16): lane holds column n=lane&15; K = {kb..kb+7, 16+kb..16+kb+7}, kb=(lane>>4)*8.
// base points at row-major [n][k] storage (i.e. W rows = output columns).
__device__ __forceinline__ bf16x16 frag_b(const __bf16* base, int ld) {
  const int lane = threadIdx.x & 31;
  const int c    = lane & 15;
  const int kb   = (lane >> 4) << 3;
  const __bf16* p = base + c * ld + kb;
  bf16x8 lo = *(const bf16x8*)(p);
  bf16x8 hi = *(const bf16x8*)(p + 16);
  bf16x16 f;
#pragma unroll
  for (int i = 0; i < 8; ++i) { f[i] = lo[i]; f[i + 8] = hi[i]; }
  return f;
}

__device__ __forceinline__ f32x8 wmma_bf16(bf16x16 a, bf16x16 b, f32x8 c) {
  return __builtin_amdgcn_wmma_f32_16x16x32_bf16(false, a, false, b, (short)0, c,
                                                 false, false);
}

// convert 16 f32 -> two bf16x8, all-in-bounds vector path
__device__ __forceinline__ void cvt16_full(const float* __restrict__ src,
                                           bf16x8 out[2]) {
#pragma unroll
  for (int q = 0; q < 2; ++q) {
    const float4 f0 = *(const float4*)(src + q * 8);
    const float4 f1 = *(const float4*)(src + q * 8 + 4);
    bf16x8 r;
    r[0] = (__bf16)f0.x; r[1] = (__bf16)f0.y;
    r[2] = (__bf16)f0.z; r[3] = (__bf16)f0.w;
    r[4] = (__bf16)f1.x; r[5] = (__bf16)f1.y;
    r[6] = (__bf16)f1.z; r[7] = (__bf16)f1.w;
    out[q] = r;
  }
}

// guarded path (only instantiated for the K=300->320 padded GEMM)
__device__ __forceinline__ void cvt16_guard(const float* __restrict__ src,
                                            int kc, int Kreal, bf16x8 out[2]) {
  if (kc + 16 <= Kreal) {
    cvt16_full(src, out);
  } else {
#pragma unroll
    for (int q = 0; q < 2; ++q) {
      bf16x8 r;
#pragma unroll
      for (int e = 0; e < 8; ++e)
        r[e] = (kc + q * 8 + e < Kreal) ? (__bf16)src[q * 8 + e] : (__bf16)0.0f;
      out[q] = r;
    }
  }
}

// ---------------------------------------------------------------------------
// Generic GEMM:  C[m,n] = (sum_k A[m,k] * W[n,k] + bias[n]) * outscale
// Template-specialized:
//   PADK  : source rows valid for k < Kreal, zero-pad to Kdim during staging
//   AF32  : A is f32 (convert while staging), else bf16
//   OUTF32: store f32 into d_out with row remap m -> b*257+1+k, else bf16
// Block: 256 thr = 8 waves (2x4), macro tile 128x128, wave tile 64x32.
// Software-pipelined: tile k+1 global loads overlap tile k WMMAs.
// ---------------------------------------------------------------------------
#define GLDT 40   // padded LDS stride (elements)

template <bool PADK, bool AF32, bool OUTF32>
__global__ __launch_bounds__(256)
void wmma_gemm_kernel(const float* __restrict__ Af, const __bf16* __restrict__ Ab,
                      int lda, int Kreal, int Kdim,
                      const float* __restrict__ W, int ldw,
                      const float* __restrict__ bias, float outscale,
                      __bf16* __restrict__ outb, float* __restrict__ outf,
                      int ldo)
{
  __shared__ __align__(16) __bf16 As[128 * GLDT];
  __shared__ __align__(16) __bf16 Ws[128 * GLDT];

  const int tid  = threadIdx.x;
  const int wid  = tid >> 5;
  const int lane = tid & 31;
  const int mbase = blockIdx.x * 128;
  const int nbase = blockIdx.y * 128;
  const int wm = (wid >> 2) * 64;   // 0 or 64
  const int wn = (wid & 3) * 32;    // 0,32,64,96

  f32x8 acc[4][2] = {};

  const int srow  = tid >> 1;        // 0..127
  const int shalf = (tid & 1) * 16;  // 0 or 16

  bf16x8 areg[2], wreg[2];

  auto load_tiles = [&](int k0) {
    const int kc = k0 + shalf;
    if (AF32) {
      const float* src = Af + (size_t)(mbase + srow) * lda + kc;
      if (PADK) cvt16_guard(src, kc, Kreal, areg);
      else      cvt16_full(src, areg);
    } else {
      const __bf16* src = Ab + (size_t)(mbase + srow) * lda + kc;
      areg[0] = *(const bf16x8*)(src);
      areg[1] = *(const bf16x8*)(src + 8);
    }
    const float* wsrc = W + (size_t)(nbase + srow) * ldw + kc;
    if (PADK) cvt16_guard(wsrc, kc, Kreal, wreg);
    else      cvt16_full(wsrc, wreg);
  };

  load_tiles(0);

  for (int k0 = 0; k0 < Kdim; k0 += 32) {
    // commit staged registers to LDS
    {
      __bf16* ad = &As[srow * GLDT + shalf];
      *(bf16x8*)(ad)     = areg[0];
      *(bf16x8*)(ad + 8) = areg[1];
      __bf16* wd = &Ws[srow * GLDT + shalf];
      *(bf16x8*)(wd)     = wreg[0];
      *(bf16x8*)(wd + 8) = wreg[1];
    }
    __syncthreads();

    // issue next tile's global loads while WMMAs run
    if (k0 + 32 < Kdim) load_tiles(k0 + 32);

    bf16x16 bfr[2];
#pragma unroll
    for (int j = 0; j < 2; ++j)
      bfr[j] = frag_b(&Ws[(wn + j * 16) * GLDT], GLDT);
#pragma unroll
    for (int i = 0; i < 4; ++i) {
      bf16x16 afr = frag_a(&As[(wm + i * 16) * GLDT], GLDT);
#pragma unroll
      for (int j = 0; j < 2; ++j)
        acc[i][j] = wmma_bf16(afr, bfr[j], acc[i][j]);
    }
    __syncthreads();
  }

  // store: lane holds col = lane&15, rows (lane>>4)*8 + r
  const int colw  = lane & 15;
  const int rbase = (lane >> 4) * 8;
#pragma unroll
  for (int i = 0; i < 4; ++i) {
#pragma unroll
    for (int j = 0; j < 2; ++j) {
      const int n = nbase + wn + j * 16 + colw;
      const float bv = bias ? bias[n] : 0.0f;
#pragma unroll
      for (int r = 0; r < 8; ++r) {
        const int m = mbase + wm + i * 16 + rbase + r;
        const float v = (acc[i][j][r] + bv) * outscale;
        if (OUTF32) {
          const size_t mo = (size_t)m + (size_t)(m >> 8) + 1;  // row remap
          outf[mo * (size_t)ldo + n] = v;
        } else {
          outb[(size_t)m * ldo + n] = (__bf16)v;
        }
      }
    }
  }
}

// ---------------------------------------------------------------------------
// Fused attention per (query-128-block, head, batch).
// Block = 256 thr = 8 waves; wave w owns query rows w*16..w*16+15 (full rows
// -> register softmax).  scores = (Q*scale) @ K^T, mask by length, softmax,
// ctx = attn @ V.   LDS: buf0 = Q(128x104) then attn(128x136),
//                        buf1 = K(128x104) then V^T(96x136).
// ---------------------------------------------------------------------------
#define LQK 104
#define LAV 136

__global__ __launch_bounds__(256)
void attn_kernel(const __bf16* __restrict__ q, const __bf16* __restrict__ k,
                 const __bf16* __restrict__ v, const int* __restrict__ lengths,
                 __bf16* __restrict__ ctx)
{
  __shared__ __align__(16) __bf16 buf0[128 * LAV];  // 34816 B
  __shared__ __align__(16) __bf16 buf1[128 * LQK];  // 26624 B

  const int qblk = blockIdx.x;   // 0..1
  const int hh   = blockIdx.y;   // 0..7
  const int b    = blockIdx.z;   // 0..255
  const int tid  = threadIdx.x;
  const int wid  = tid >> 5;
  const int lane = tid & 31;

  int len = lengths[b];
  len = min(max(len, 1), NN);

  __bf16* Qs = buf0;  __bf16* Atn = buf0;
  __bf16* Ks = buf1;  __bf16* Vt  = buf1;

  // load Q (128 x 96) and K (128 x 96), 12 chunks of 8 per row
  for (int c = tid; c < 128 * 12; c += 256) {
    const int r = c / 12, cc = (c % 12) * 8;
    *(bf16x8*)&Qs[r * LQK + cc] =
        *(const bf16x8*)&q[(size_t)(qblk * 128 + r) * DD + hh * DH + cc];
  }
  for (int c = tid; c < 128 * 12; c += 256) {
    const int r = c / 12, cc = (c % 12) * 8;
    *(bf16x8*)&Ks[r * LQK + cc] =
        *(const bf16x8*)&k[(size_t)(b * NN + r) * DD + hh * DH + cc];
  }
  __syncthreads();

  // scores: wave rows x all 128 tokens  (8 tiles of 16 cols)
  f32x8 sc[8] = {};
  for (int kk = 0; kk < DH; kk += 32) {
    bf16x16 a = frag_a(&Qs[(wid * 16) * LQK + kk], LQK);
#pragma unroll
    for (int t = 0; t < 8; ++t) {
      bf16x16 bb = frag_b(&Ks[(t * 16) * LQK + kk], LQK);
      sc[t] = wmma_bf16(a, bb, sc[t]);
    }
  }
  __syncthreads();  // Qs/Ks free now

  // stage V^T (96 dh x 128 tok) into buf1 — vectorized global reads
  for (int i = tid; i < (DH / 8) * NN; i += 256) {   // 1536 chunks
    const int n   = i & 127;
    const int dhb = i >> 7;                          // 0..11
    const bf16x8 vv =
        *(const bf16x8*)&v[(size_t)(b * NN + n) * DD + hh * DH + dhb * 8];
#pragma unroll
    for (int e = 0; e < 8; ++e)
      Vt[(dhb * 8 + e) * LAV + n] = vv[e];
  }

  // register softmax: lane holds col = lane&15 of tile t; rows (lane>>4)*8+r
  const int colw = lane & 15;
  const int rsel = (lane >> 4) * 8;
#pragma unroll
  for (int r = 0; r < 8; ++r) {
    float vals[8];
    float mx = NEGV;
#pragma unroll
    for (int t = 0; t < 8; ++t) {
      const int tok = t * 16 + colw;
      const float s = (tok < len) ? sc[t][r] : NEGV;
      vals[t] = s;
      mx = fmaxf(mx, s);
    }
#pragma unroll
    for (int mk = 1; mk < 16; mk <<= 1) mx = fmaxf(mx, __shfl_xor(mx, mk, 32));
    float sum = 0.0f;
#pragma unroll
    for (int t = 0; t < 8; ++t) { vals[t] = __expf(vals[t] - mx); sum += vals[t]; }
#pragma unroll
    for (int mk = 1; mk < 16; mk <<= 1) sum += __shfl_xor(sum, mk, 32);
    const float inv = 1.0f / sum;
#pragma unroll
    for (int t = 0; t < 8; ++t) sc[t][r] = vals[t] * inv;
  }

  // attn -> buf0 (bf16)
  const int rowl = wid * 16 + rsel;
#pragma unroll
  for (int t = 0; t < 8; ++t) {
    const int cc = t * 16 + colw;
#pragma unroll
    for (int r = 0; r < 8; ++r)
      Atn[(rowl + r) * LAV + cc] = (__bf16)sc[t][r];
  }
  __syncthreads();

  // ctx = attn(16x128) @ V(128x96): 6 n-tiles, 4 k-steps
  f32x8 co[6] = {};
  for (int kk = 0; kk < NN; kk += 32) {
    bf16x16 a = frag_a(&Atn[(wid * 16) * LAV + kk], LAV);
#pragma unroll
    for (int j = 0; j < 6; ++j) {
      bf16x16 bb = frag_b(&Vt[(j * 16) * LAV + kk], LAV);
      co[j] = wmma_bf16(a, bb, co[j]);
    }
  }

  // store ctx [B, KQ, D] with D col = h*DH + dh
#pragma unroll
  for (int j = 0; j < 6; ++j) {
    const int dh = j * 16 + colw;
#pragma unroll
    for (int r = 0; r < 8; ++r) {
      const int qi = qblk * 128 + wid * 16 + rsel + r;
      ctx[((size_t)b * KQ + qi) * DD + hh * DH + dh] = (__bf16)co[j][r];
    }
  }
}

// ---------------------------------------------------------------------------
// CLS row: masked mean of h over nodes -> out_tokens[b, 0, :]
// 96 threads, each owns 8 contiguous channels (b128 loads of h).
// ---------------------------------------------------------------------------
__global__ __launch_bounds__(96)
void cls_kernel(const __bf16* __restrict__ h, const int* __restrict__ lengths,
                float* __restrict__ out)
{
  const int b = blockIdx.x;
  int len = lengths[b];
  len = min(max(len, 1), NN);
  const int c = threadIdx.x;  // 0..95, chunk of 8 channels
  float s[8] = {};
  for (int n = 0; n < len; ++n) {
    const bf16x8 hv = *(const bf16x8*)&h[((size_t)b * NN + n) * DD + c * 8];
#pragma unroll
    for (int e = 0; e < 8; ++e) s[e] += (float)hv[e];
  }
  const float inv = 1.0f / (float)len;
  float* o = &out[(size_t)b * (KQ + 1) * DD + c * 8];
#pragma unroll
  for (int e = 0; e < 8; ++e) o[e] = s[e] * inv;
}

__global__ __launch_bounds__(256)
void mask_kernel(float* __restrict__ out)
{
  const size_t base = (size_t)BB * (KQ + 1) * DD;
  const int i = blockIdx.x * 256 + threadIdx.x;
  if (i < BB * (KQ + 1)) out[base + i] = 1.0f;
}

// ---------------------------------------------------------------------------
extern "C" void kernel_launch(void* const* d_in, const int* in_sizes, int n_in,
                              void* d_out, int out_size, void* d_ws, size_t ws_size,
                              hipStream_t stream)
{
  (void)in_sizes; (void)n_in; (void)out_size; (void)ws_size;
  const float* x         = (const float*)d_in[0];
  const int*   lengths   = (const int*)  d_in[1];
  const float* proj_w    = (const float*)d_in[2];
  const float* query_tok = (const float*)d_in[3];
  const float* in_proj_w = (const float*)d_in[4];
  const float* in_proj_b = (const float*)d_in[5];
  const float* out_w     = (const float*)d_in[6];
  const float* out_b     = (const float*)d_in[7];
  float* out = (float*)d_out;

  // workspace carve-out (256B aligned)
  char* ws = (char*)d_ws;
  size_t off = 0;
  auto carve = [&](size_t bytes) {
    void* p = ws + off;
    off += (bytes + 255) & ~(size_t)255;
    return p;
  };
  __bf16* h_bf = (__bf16*)carve((size_t)BB * NN * DD * 2);  // 50.3 MB
  __bf16* q_bf = (__bf16*)carve((size_t)KQ * DD * 2);
  __bf16* k_bf = (__bf16*)carve((size_t)BB * NN * DD * 2);
  __bf16* v_bf = (__bf16*)carve((size_t)BB * NN * DD * 2);
  __bf16* c_bf = (__bf16*)carve((size_t)BB * KQ * DD * 2);  // 100.7 MB

  const float qscale = 0.1020620726159658f;  // 1/sqrt(96)

  // h = x @ proj_w^T          [32768,300pad320] x [768,300pad320]
  wmma_gemm_kernel<true, true, false>
      <<<dim3(BB * NN / 128, DD / 128), 256, 0, stream>>>(
          x, nullptr, DIN, DIN, 320, proj_w, DIN, nullptr, 1.0f,
          h_bf, nullptr, DD);

  // cls row (reads h)
  cls_kernel<<<BB, 96, 0, stream>>>(h_bf, lengths, out);

  // q = (query_tok @ wq^T + bq) * scale     [256,768]
  wmma_gemm_kernel<false, true, false>
      <<<dim3(KQ / 128, DD / 128), 256, 0, stream>>>(
          query_tok, nullptr, DD, DD, DD, in_proj_w, DD, in_proj_b, qscale,
          q_bf, nullptr, DD);

  // k = h @ wk^T + bk
  wmma_gemm_kernel<false, false, false>
      <<<dim3(BB * NN / 128, DD / 128), 256, 0, stream>>>(
          nullptr, h_bf, DD, DD, DD, in_proj_w + (size_t)DD * DD, DD,
          in_proj_b + DD, 1.0f, k_bf, nullptr, DD);

  // v = h @ wv^T + bv
  wmma_gemm_kernel<false, false, false>
      <<<dim3(BB * NN / 128, DD / 128), 256, 0, stream>>>(
          nullptr, h_bf, DD, DD, DD, in_proj_w + (size_t)2 * DD * DD, DD,
          in_proj_b + 2 * DD, 1.0f, v_bf, nullptr, DD);

  // fused attention -> ctx [B, KQ, D]
  attn_kernel<<<dim3(KQ / 128, HH, BB), 256, 0, stream>>>(
      q_bf, k_bf, v_bf, lengths, c_bf);

  // patches = ctx @ out_w^T + out_b -> out_tokens rows 1..256 per batch
  wmma_gemm_kernel<false, false, true>
      <<<dim3(BB * KQ / 128, DD / 128), 256, 0, stream>>>(
          nullptr, c_bf, DD, DD, DD, out_w, DD, out_b, 1.0f,
          nullptr, out, DD);

  // attn_mask = ones
  mask_kernel<<<(BB * (KQ + 1) + 255) / 256, 256, 0, stream>>>(out);
}
